// HierarchicalTransformer_76201309765910
// MI455X (gfx1250) — compile-verified, matrix operand order checked by reading, not run on hardware
//
#include <hip/hip_runtime.h>
#include <hip/hip_bf16.h>

typedef __attribute__((ext_vector_type(16))) __bf16 v16bf;
typedef __attribute__((ext_vector_type(8)))  __bf16 v8bf;
typedef __attribute__((ext_vector_type(8)))  float  v8f;

#define DMODEL 256
#define NHEAD  8
#define DHEAD  32
#define DFFN   2048

// ---------------------------------------------------------------------------
// fp32 -> bf16 conversion / fp32 copy (grid-stride)
// ---------------------------------------------------------------------------
__global__ void k_cvt_bf16(__bf16* __restrict__ dst, const float* __restrict__ src, size_t n) {
    size_t i = (size_t)blockIdx.x * blockDim.x + threadIdx.x;
    size_t stride = (size_t)gridDim.x * blockDim.x;
    for (; i < n; i += stride) dst[i] = (__bf16)src[i];
}

__global__ void k_copy_f32(float* __restrict__ dst, const float* __restrict__ src, size_t n) {
    size_t i = (size_t)blockIdx.x * blockDim.x + threadIdx.x;
    size_t stride = (size_t)gridDim.x * blockDim.x;
    for (; i < n; i += stride) dst[i] = src[i];
}

// ---------------------------------------------------------------------------
// GEMM: C[M,N] = A[M,K] @ W[N,K]^T + bias, bf16 inputs, fp32 accumulate.
// Block = 256 threads = 8 waves covering 128(M) x 64(N). The 64x32 B tile is
// staged once per block into double-buffered LDS through a 2-stage register
// pipeline: the global fetch for step k+1 is issued at step k and only
// ds_store'd (and waited on) at step k+1, so each wave has a full K-step of
// WMMA + ds_load work covering the global latency. global_prefetch_b8 warms
// the W/A streams two K-steps ahead. LDS rows padded to 80B for conflict-free
// ds_load_b128 fragment reads; all four B fragments are loaded before the
// WMMA burst so the DS waits can cascade. 4 WMMAs per wave per K-step.
// Grid = (ceil(M/128), N/64). N % 64 == 0, K % 32 == 0.
// ---------------------------------------------------------------------------
#define BSTRIDE 40  // halfs per B row in LDS (32 data + 8 pad) = 80 bytes

template <bool RELU, bool OUTBF16>
__global__ __launch_bounds__(256) void k_gemm_bf16(
    const __bf16* __restrict__ A, const __bf16* __restrict__ W,
    const float* __restrict__ bias, float* __restrict__ Cf,
    __bf16* __restrict__ Cb, int M, int N, int K)
{
    __shared__ __bf16 lb[2][64 * BSTRIDE];

    const int tid   = threadIdx.x;
    const int lane  = tid & 31;
    const int wave  = tid >> 5;
    const int tileM = blockIdx.x * 128 + wave * 16;
    const int nbase = blockIdx.y * 64;
    const int l15   = lane & 15;
    const int kh8   = (lane >> 4) * 8;   // upper half-wave: K+8 / K+24

    int ar = tileM + l15; if (ar >= M) ar = M - 1;   // clamp M edge
    const __bf16* arow = A + (size_t)ar * K + kh8;

    // cooperative B staging: thread t -> row t>>2 (0..63), 16B chunk t&3
    const int srow = tid >> 2, schk = tid & 3;
    const __bf16* gsrc = W + (size_t)(nbase + srow) * K + schk * 8;
    __bf16* ldst0 = &lb[0][srow * BSTRIDE + schk * 8];
    __bf16* ldst1 = &lb[1][srow * BSTRIDE + schk * 8];

    // pipeline prologue: fetch k=0 staging chunk + k=0 A fragment
    v8bf bReg = *(const v8bf*)(gsrc);
    v8bf aLo  = *(const v8bf*)(arow);
    v8bf aHi  = *(const v8bf*)(arow + 16);

    v8f acc[4] = {};
    for (int k0 = 0; k0 < K; k0 += 32) {
        const int cur = (k0 >> 5) & 1;
        const __bf16* lcur = &lb[cur][0];
        const bool more = (k0 + 32) < K;

        // commit the chunk fetched one step ago, make tile visible
        *(v8bf*)(cur ? ldst1 : ldst0) = bReg;
        __syncthreads();

        if ((k0 + 64) < K) {                       // prefetch tile after next
            __builtin_prefetch(gsrc + k0 + 64, 0, 3);
            __builtin_prefetch(arow + k0 + 64, 0, 3);
        }

        v8bf aLoN = aLo, aHiN = aHi;
        if (more) {                                // next-step fetches in flight
            bReg = *(const v8bf*)(gsrc + k0 + 32);
            aLoN = *(const v8bf*)(arow + k0 + 32);
            aHiN = *(const v8bf*)(arow + k0 + 48);
        }

        v16bf a = __builtin_shufflevector(aLo, aHi,
                   0,1,2,3,4,5,6,7,8,9,10,11,12,13,14,15);

        // issue all 8 ds_load_b128 first, then burst the 4 WMMAs
        v16bf bf[4];
        #pragma unroll
        for (int j = 0; j < 4; ++j) {
            const __bf16* bp = lcur + (size_t)(16 * j + l15) * BSTRIDE + kh8;
            v8bf blo = *(const v8bf*)bp;
            v8bf bhi = *(const v8bf*)(bp + 16);
            bf[j] = __builtin_shufflevector(blo, bhi,
                     0,1,2,3,4,5,6,7,8,9,10,11,12,13,14,15);
        }
        #pragma unroll
        for (int j = 0; j < 4; ++j)
            acc[j] = __builtin_amdgcn_wmma_f32_16x16x32_bf16(
                         false, a, false, bf[j], (short)0, acc[j], false, false);

        aLo = aLoN; aHi = aHiN;
    }

    const int rbase = tileM + ((lane >> 4) << 3);  // M = r (lanes 0-15) / 8+r
    #pragma unroll
    for (int j = 0; j < 4; ++j) {
        const int col = nbase + 16 * j + l15;
        const float bb = bias ? bias[col] : 0.0f;
        #pragma unroll
        for (int r = 0; r < 8; ++r) {
            const int row = rbase + r;
            if (row < M) {
                float v = acc[j][r] + bb;
                if (RELU) v = fmaxf(v, 0.0f);
                if (OUTBF16) Cb[(size_t)row * N + col] = (__bf16)v;
                else         Cf[(size_t)row * N + col] = v;
            }
        }
    }
}

// ---------------------------------------------------------------------------
// Attention: qkv [B,S,3*D] (cols 0..255=q, 256..511=k, 512..767=v; head h at
// h*32). One thread per (query, head): online softmax over keys, with key
// padding mask j > lengths[b] (matches reference -1e9 masking).
// ---------------------------------------------------------------------------
__global__ void k_attn(const float* __restrict__ qkv, float* __restrict__ out,
                       const int* __restrict__ lengths, int B, int S)
{
    const int bh = blockIdx.y;
    const int b = bh >> 3, h = bh & 7;
    const int q = blockIdx.x * blockDim.x + threadIdx.x;
    if (q >= S) return;
    const float scale = 0.1767766952966369f;  // 1/sqrt(32)
    const float* base = qkv + (size_t)b * S * (3 * DMODEL);
    const float* qp = base + (size_t)q * (3 * DMODEL) + h * DHEAD;
    float qv[DHEAD];
    #pragma unroll
    for (int d = 0; d < DHEAD; ++d) qv[d] = qp[d] * scale;

    int jmax = S - 1;
    if (lengths) { int len = lengths[b]; if (len < jmax) jmax = len; }

    float m = -3.0e38f, ssum = 0.0f, accv[DHEAD] = {};
    for (int j = 0; j <= jmax; ++j) {
        const float* kp = base + (size_t)j * (3 * DMODEL) + DMODEL + h * DHEAD;
        float s = 0.0f;
        #pragma unroll
        for (int d = 0; d < DHEAD; ++d) s += qv[d] * kp[d];
        const float mn = fmaxf(m, s);
        const float corr = __expf(m - mn);
        const float w = __expf(s - mn);
        ssum = ssum * corr + w;
        const float* vp = base + (size_t)j * (3 * DMODEL) + 2 * DMODEL + h * DHEAD;
        #pragma unroll
        for (int d = 0; d < DHEAD; ++d) accv[d] = accv[d] * corr + w * vp[d];
        m = mn;
    }
    const float inv = 1.0f / ssum;
    float* op = out + ((size_t)b * S + q) * DMODEL + h * DHEAD;
    #pragma unroll
    for (int d = 0; d < DHEAD; ++d) op[d] = accv[d] * inv;
}

// ---------------------------------------------------------------------------
// Residual + LayerNorm (post-norm): out = LN(x + y) * w + b. One row / block.
// ---------------------------------------------------------------------------
__global__ __launch_bounds__(256) void k_addln(
    const float* __restrict__ x, const float* __restrict__ y,
    const float* __restrict__ w, const float* __restrict__ b,
    float* __restrict__ out, int rows)
{
    const int row = blockIdx.x, t = threadIdx.x;
    const float v = x[(size_t)row * DMODEL + t] + y[(size_t)row * DMODEL + t];
    __shared__ float red[256];
    red[t] = v; __syncthreads();
    for (int s = 128; s > 0; s >>= 1) { if (t < s) red[t] += red[t + s]; __syncthreads(); }
    const float mean = red[0] * (1.0f / DMODEL);
    __syncthreads();
    const float dv = v - mean;
    red[t] = dv * dv; __syncthreads();
    for (int s = 128; s > 0; s >>= 1) { if (t < s) red[t] += red[t + s]; __syncthreads(); }
    const float var = red[0] * (1.0f / DMODEL);
    out[(size_t)row * DMODEL + t] = dv * rsqrtf(var + 1e-5f) * w[t] + b[t];
}

// ---------------------------------------------------------------------------
// Sequence assembly kernels
// ---------------------------------------------------------------------------
__global__ void k_hopseq(float* __restrict__ dst, const float* __restrict__ tok,
                         const float* __restrict__ cls, const int* __restrict__ hlist)
{
    const int hop = blockIdx.x / 17, s = blockIdx.x % 17, t = threadIdx.x;
    float v;
    if (s == 0) v = cls[t];
    else        v = tok[(size_t)hlist[hop * 16 + s - 1] * DMODEL + t];
    dst[(size_t)blockIdx.x * DMODEL + t] = v;
}

__global__ void k_pos(const int* __restrict__ lev, int* __restrict__ pos,
                      int* __restrict__ counts)
{
    const int i = threadIdx.x;          // <<<1,512>>>
    const int li = lev[i];
    int c = 0;
    for (int j = 0; j < i; ++j) c += (lev[j] == li);
    pos[i] = c;
    if (i < 16) {
        int cc = 0;
        for (int j = 0; j < 512; ++j) cc += (lev[j] == i);
        counts[i] = cc;
    }
}

__global__ void k_levinit(float* __restrict__ dst, const float* __restrict__ cls) {
    const int s = blockIdx.x % 33, t = threadIdx.x;
    dst[(size_t)blockIdx.x * DMODEL + t] = (s == 0) ? cls[t] : 0.0f;
}

__global__ void k_levscatter(float* __restrict__ grid, const float* __restrict__ htok,
                             const int* __restrict__ lev, const int* __restrict__ pos)
{
    const int i = blockIdx.x, t = threadIdx.x;
    grid[((size_t)lev[i] * 33 + 1 + pos[i]) * DMODEL + t] = htok[(size_t)i * DMODEL + t];
}

__global__ void k_graphseq(float* __restrict__ dst, const float* __restrict__ cls,
                           const float* __restrict__ lt)
{
    const int s = blockIdx.x, t = threadIdx.x;
    dst[(size_t)s * DMODEL + t] = (s == 0) ? cls[t] : lt[(size_t)(s - 1) * DMODEL + t];
}

__global__ void k_extract_cls(float* __restrict__ dst, const float* __restrict__ src, int S) {
    dst[(size_t)blockIdx.x * DMODEL + threadIdx.x] =
        src[(size_t)blockIdx.x * S * DMODEL + threadIdx.x];
}

// ---------------------------------------------------------------------------
// Host orchestration
// ---------------------------------------------------------------------------
extern "C" void kernel_launch(void* const* d_in, const int* in_sizes, int n_in,
                              void* d_out, int out_size, void* d_ws, size_t ws_size,
                              hipStream_t stream)
{
    const float* tokens      = (const float*)d_in[0];
    const float* masked      = (const float*)d_in[1];
    const float* hop_nodes   = (const float*)d_in[2];
    const float* level_nodes = (const float*)d_in[3];
    const float* graph_nodes = (const float*)d_in[4];
    const float* in_w  = (const float*)d_in[5];
    const float* in_b  = (const float*)d_in[6];
    const float* out_w = (const float*)d_in[7];
    const float* out_b = (const float*)d_in[8];
    const float* f1w   = (const float*)d_in[9];
    const float* f1b   = (const float*)d_in[10];
    const float* f2w   = (const float*)d_in[11];
    const float* f2b   = (const float*)d_in[12];
    const float* l1w   = (const float*)d_in[13];
    const float* l1b   = (const float*)d_in[14];
    const float* l2w   = (const float*)d_in[15];
    const float* l2b   = (const float*)d_in[16];
    const int* hop_list = (const int*)d_in[17];
    const int* hop_lev  = (const int*)d_in[18];
    const int* hop_len  = (const int*)d_in[19];
    (void)in_sizes; (void)n_in; (void)ws_size; (void)out_size;

    // ---- bump allocator over d_ws ----
    char* wsp = (char*)d_ws;
    auto alloc = [&](size_t bytes) -> void* {
        void* r = (void*)wsp;
        wsp += (bytes + 255) & ~(size_t)255;
        return r;
    };

    const size_t NEL = 13 * 2;  // (encoder, layer) pairs
    __bf16* wb_in  = (__bf16*)alloc(NEL * 768  * 256  * 2);
    __bf16* wb_out = (__bf16*)alloc(NEL * 256  * 256  * 2);
    __bf16* wb_f1  = (__bf16*)alloc(NEL * 2048 * 256  * 2);
    __bf16* wb_f2  = (__bf16*)alloc(NEL * 256  * 2048 * 2);

    const int RMAX = 512 * 17;  // largest row count (hop encoders)
    float*  x     = (float*) alloc((size_t)RMAX * 256  * 4);
    __bf16* xb    = (__bf16*)alloc((size_t)RMAX * 256  * 2);
    float*  qkv   = (float*) alloc((size_t)RMAX * 768  * 4);
    float*  attn  = (float*) alloc((size_t)RMAX * 256  * 4);
    __bf16* attnb = (__bf16*)alloc((size_t)RMAX * 256  * 2);
    float*  o     = (float*) alloc((size_t)RMAX * 256  * 4);
    float*  x1    = (float*) alloc((size_t)RMAX * 256  * 4);
    __bf16* hb    = (__bf16*)alloc((size_t)RMAX * 2048 * 2);
    float*  ffo   = (float*) alloc((size_t)RMAX * 256  * 4);
    float*  hop_tok = (float*)alloc((size_t)3 * 512 * 256 * 4);
    float*  lev_tok = (float*)alloc((size_t)3 * 16  * 256 * 4);
    float*  g_tok   = (float*)alloc((size_t)3 * 256 * 4);
    int*    pos     = (int*)alloc(512 * 4);
    int*    counts  = (int*)alloc(16 * 4);

    auto cvt = [&](__bf16* d, const float* s, size_t n) {
        int blocks = (int)((n + 255) / 256);
        if (blocks > 4096) blocks = 4096;
        k_cvt_bf16<<<blocks, 256, 0, stream>>>(d, s, n);
    };
    auto cpy = [&](float* d, const float* s, size_t n) {
        int blocks = (int)((n + 255) / 256);
        if (blocks > 4096) blocks = 4096;
        k_copy_f32<<<blocks, 256, 0, stream>>>(d, s, n);
    };

    // ---- convert all weights to bf16 (L2-resident: 68 MB < 192 MB) ----
    cvt(wb_in,  in_w,  NEL * 768  * 256);
    cvt(wb_out, out_w, NEL * 256  * 256);
    cvt(wb_f1,  f1w,   NEL * 2048 * 256);
    cvt(wb_f2,  f2w,   NEL * 256  * 2048);

    // ---- one transformer encoder layer ----
    auto layer = [&](float* X, int rows, int B, int S, const int* lengths,
                     int eid, int l) {
        const size_t wi = (size_t)(eid * 2 + l);
        const __bf16* Win  = wb_in  + wi * 768  * 256;
        const __bf16* Wout = wb_out + wi * 256  * 256;
        const __bf16* W1   = wb_f1  + wi * 2048 * 256;
        const __bf16* W2   = wb_f2  + wi * 256  * 2048;
        const float* Bin  = in_b  + wi * 768;
        const float* Bout = out_b + wi * 256;
        const float* B1   = f1b   + wi * 2048;
        const float* B2   = f2b   + wi * 256;
        const int gm = (rows + 127) / 128;

        cvt(xb, X, (size_t)rows * 256);
        k_gemm_bf16<false, false><<<dim3(gm, 768 / 64), 256, 0, stream>>>(
            xb, Win, Bin, qkv, nullptr, rows, 768, 256);
        k_attn<<<dim3((S + 127) / 128, B * 8), 128, 0, stream>>>(qkv, attn, lengths, B, S);
        cvt(attnb, attn, (size_t)rows * 256);
        k_gemm_bf16<false, false><<<dim3(gm, 256 / 64), 256, 0, stream>>>(
            attnb, Wout, Bout, o, nullptr, rows, 256, 256);
        k_addln<<<rows, 256, 0, stream>>>(X, o, l1w + wi * 256, l1b + wi * 256, x1, rows);
        cvt(xb, x1, (size_t)rows * 256);
        k_gemm_bf16<true, true><<<dim3(gm, 2048 / 64), 256, 0, stream>>>(
            xb, W1, B1, nullptr, hb, rows, 2048, 256);
        k_gemm_bf16<false, false><<<dim3(gm, 256 / 64), 256, 0, stream>>>(
            hb, W2, B2, ffo, nullptr, rows, 256, 2048);
        k_addln<<<rows, 256, 0, stream>>>(x1, ffo, l2w + wi * 256, l2b + wi * 256, X, rows);
    };

    // ---- three modalities ----
    for (int m = 0; m < 3; ++m) {
        const int mk = m + 1;
        // hop encoder: 512 seqs of len 17
        k_hopseq<<<512 * 17, 256, 0, stream>>>(
            x, tokens + (size_t)mk * 8192 * 256, hop_nodes,
            hop_list + (size_t)m * 512 * 16);
        for (int l = 0; l < 2; ++l)
            layer(x, 512 * 17, 512, 17, hop_len + m * 512, mk, l);
        k_extract_cls<<<512, 256, 0, stream>>>(hop_tok + (size_t)m * 512 * 256, x, 17);

        // level encoder: pack hops by level, 16 seqs of len 33
        k_pos<<<1, 512, 0, stream>>>(hop_lev + m * 512, pos, counts);
        k_levinit<<<16 * 33, 256, 0, stream>>>(x, level_nodes);
        k_levscatter<<<512, 256, 0, stream>>>(
            x, hop_tok + (size_t)m * 512 * 256, hop_lev + m * 512, pos);
        for (int l = 0; l < 2; ++l)
            layer(x, 16 * 33, 16, 33, counts, 4 + mk, l);
        k_extract_cls<<<16, 256, 0, stream>>>(lev_tok + (size_t)m * 16 * 256, x, 33);

        // graph encoder: 1 seq of len 17
        k_graphseq<<<17, 256, 0, stream>>>(x, graph_nodes, lev_tok + (size_t)m * 16 * 256);
        for (int l = 0; l < 2; ++l)
            layer(x, 17, 1, 17, nullptr, 8 + mk, l);
        k_extract_cls<<<1, 256, 0, stream>>>(g_tok + (size_t)m * 256, x, 17);
    }

    // ---- MCM encoder over 512 masked + 3*(512+16+1) = 2099 tokens ----
    const int MCM_S = 512 + 3 * (512 + 16 + 1);
    cpy(x, masked, (size_t)512 * 256);
    for (int m = 0; m < 3; ++m) {
        float* base = x + (size_t)(512 + m * 529) * 256;
        cpy(base,             hop_tok + (size_t)m * 512 * 256, (size_t)512 * 256);
        cpy(base + 512 * 256, lev_tok + (size_t)m * 16 * 256,  (size_t)16 * 256);
        cpy(base + 528 * 256, g_tok   + (size_t)m * 256,       (size_t)256);
    }
    for (int l = 0; l < 2; ++l)
        layer(x, MCM_S, 1, MCM_S, nullptr, 12, l);

    // output: first 512 rows
    cpy((float*)d_out, x, (size_t)512 * 256);
}
